// SliceAwareAttention_41197326303868
// MI455X (gfx1250) — matrix operand order repaired and checked by reading, not compiled, not run on hardware
//
#include <hip/hip_runtime.h>
#include <hip/hip_bf16.h>

typedef __attribute__((ext_vector_type(16))) __bf16 v16bf;
typedef __attribute__((ext_vector_type(8)))  __bf16 v8bf;
typedef __attribute__((ext_vector_type(8)))  float  v8f;

#define HEADS 12
#define SEQ   2048
#define CDIM  768
#define DHEAD 64
#define BATCH 2
#define QKV_N (3 * CDIM)

// LDS tile row pitches (padded for bank-conflict-free 16B fragment reads)
#define KP 72   // K tile row pitch (elems): 128B row + 16B pad
#define VP 40   // V tile row pitch (elems): 64B row + 16B pad

#if __has_builtin(__builtin_amdgcn_tensor_load_to_lds)
#define USE_TDM 1
#else
#define USE_TDM 0
#endif

// ---------------------------------------------------------------------------
// WMMA fragment helpers (bf16 16x16x32); A/B use contiguous k order, same in both.
// A (16x32): lane L holds row (L&15); elems 0..15 = K[(L>>4)*16 .. +16)
// B (32x16): lane L holds col (L&15); elems 0..15 = K[(L>>4)*16 .. +16)
// C/D (16x16 f32): lane L col (L&15); elem e -> row e + 8*(L>>4)
// ---------------------------------------------------------------------------
static __device__ inline v16bf concat8(v8bf lo, v8bf hi) {
  v16bf r;
#pragma unroll
  for (int i = 0; i < 8; ++i) { r[i] = lo[i]; r[i + 8] = hi[i]; }
  return r;
}

static __device__ inline v16bf load_a16(const __bf16* base, int ld) {
  const int lane = threadIdx.x & 31;
  const int r = lane & 15, hi = lane >> 4;
  const __bf16* p = base + (size_t)r * ld + hi * 16;
  v8bf lo = *(const v8bf*)p;
  v8bf hh = *(const v8bf*)(p + 8);
  return concat8(lo, hh);
}

static __device__ inline v16bf load_b16(const __bf16* base, int ld) {
  const int lane = threadIdx.x & 31;
  const int c = lane & 15, hi = lane >> 4;
  const __bf16* p = base + (size_t)c * ld + hi * 16;
  v8bf lo = *(const v8bf*)p;
  v8bf hh = *(const v8bf*)(p + 8);
  return concat8(lo, hh);
}

static __device__ inline v8f wmma_bf16(v16bf a, v16bf b, v8f c) {
  return __builtin_amdgcn_wmma_f32_16x16x32_bf16(false, a, false, b, (short)0, c,
                                                 false, false);
}

// ---------------------------------------------------------------------------
// Generic pointers to __shared__ carry the LDS byte offset in addr[31:0].
// ---------------------------------------------------------------------------
static __device__ inline unsigned lds_off32(const void* p) {
  return (unsigned)(unsigned long long)(uintptr_t)p;
}

// ---------------------------------------------------------------------------
// CDNA5 Tensor Data Mover: build a D# (groups 0/1 per ISA 8.3/8.4) and issue
// TENSOR_LOAD_TO_LDS.  data_size=2 (4-byte units).  Padding applied on LDS
// writes implements the banked tile pitches in hardware.
// ---------------------------------------------------------------------------
#if USE_TDM
typedef unsigned int tdm_v4u __attribute__((ext_vector_type(4)));
typedef int          tdm_v8i __attribute__((ext_vector_type(8)));
typedef int          tdm_v4i __attribute__((ext_vector_type(4)));

static __device__ inline void tdm_load_2d(unsigned lds_addr, unsigned long long ga,
                                          int d0, int d1, int d2, int d3, int d4,
                                          int d5) {
  tdm_v4u g0;
  g0[0] = 1u;                                   // count=1, user-mode, no gather
  g0[1] = lds_addr;                             // LDS byte address
  g0[2] = (unsigned)ga;                         // global_addr[31:0]
  g0[3] = (unsigned)(ga >> 32) | (2u << 30);    // global_addr[56:32] | type=2
  tdm_v8i g1;
  g1[0] = d0;  // wg_mask=0 | data_size<<16 | pad_enable<<20 | pad_itv<<22 | pad_amt<<25
  g1[1] = d1;  // atomic_barrier_addr=0 | tensor_dim0[15:0]<<16
  g1[2] = d2;  // tensor_dim0[31:16] | tensor_dim1[15:0]<<16
  g1[3] = d3;  // tensor_dim1[31:16] | tile_dim0<<16
  g1[4] = d4;  // tile_dim1 | tile_dim2<<16
  g1[5] = d5;  // tensor_dim0_stride[31:0]
  g1[6] = 0;   // tensor_dim0_stride[47:32] | tensor_dim1_stride[15:0]<<16
  g1[7] = 0;   // tensor_dim1_stride[47:16]
  tdm_v4i gz = {0, 0, 0, 0};
#if __clang_major__ >= 23
  tdm_v8i gz8 = {0, 0, 0, 0, 0, 0, 0, 0};
  __builtin_amdgcn_tensor_load_to_lds(g0, g1, gz, gz, gz8, 0);
#else
  __builtin_amdgcn_tensor_load_to_lds(g0, g1, gz, gz, 0);
#endif
}

// K tile: 1D stream, 1024 dwords; pad 4 dwords after every 32 dwords (128B).
static __device__ inline void tdm_load_ktile(unsigned lds_addr, unsigned long long ga) {
  tdm_load_2d(lds_addr, ga,
              (2 << 16) | (1 << 20) | (4 << 22) | (3 << 25),
              1024 << 16,   // tensor_dim0 = 1024
              1 << 16,      // tensor_dim1 = 1
              1024 << 16,   // tile_dim0 = 1024 dwords
              1,            // tile_dim1 = 1
              1024);        // row stride (unused for 1 row)
}

// V tile: 2D, 16 dwords x 64 rows, global row stride 1024 dwords;
// pad 4 dwords after every 16 dwords (64B).
static __device__ inline void tdm_load_vtile(unsigned lds_addr, unsigned long long ga) {
  tdm_load_2d(lds_addr, ga,
              (2 << 16) | (1 << 20) | (3 << 22) | (3 << 25),
              1024 << 16,   // tensor_dim0 = 1024 (full row, no OOB)
              64 << 16,     // tensor_dim1 = 64
              16 << 16,     // tile_dim0 = 16 dwords (64B)
              64,           // tile_dim1 = 64 rows
              1024);        // tensor_dim0_stride = 1024 dwords
}
#else
// Fallback: per-lane async global->LDS copies (ASYNCcnt-tracked).
static __device__ inline void async_b128(unsigned lds_off, const void* gptr) {
  unsigned long long ga = (unsigned long long)(uintptr_t)gptr;
  asm volatile("global_load_async_to_lds_b128 %0, %1, off"
               :
               : "v"(lds_off), "v"(ga)
               : "memory");
}
#endif

static __device__ inline void wait_stage_pipe() {  // <= one tile in flight
#if USE_TDM
  __builtin_amdgcn_s_wait_tensorcnt(2);
#else
  asm volatile("s_wait_asynccnt 0x4" ::: "memory");
#endif
}
static __device__ inline void wait_stage_all() {
#if USE_TDM
  __builtin_amdgcn_s_wait_tensorcnt(0);
#else
  asm volatile("s_wait_asynccnt 0x0" ::: "memory");
#endif
}

// ---------------------------------------------------------------------------
// f32 -> bf16 conversion (grid-stride)
// ---------------------------------------------------------------------------
__global__ __launch_bounds__(256) void f32_to_bf16_k(const float* __restrict__ src,
                                                     __bf16* __restrict__ dst, int n) {
  int i = blockIdx.x * blockDim.x + threadIdx.x;
  int stride = gridDim.x * blockDim.x;
  for (; i < n; i += stride) dst[i] = (__bf16)src[i];
}

// ---------------------------------------------------------------------------
// QKV GEMM: [4096 x 768] x [768 x 2304]^T + bias
// Block 256 thr = 8 waves (4 M x 2 N); WG tile 128x128; wave tile 32x64.
// ---------------------------------------------------------------------------
__global__ __launch_bounds__(256) void qkv_gemm_k(const __bf16* __restrict__ X,
                                                  const __bf16* __restrict__ W,
                                                  const float* __restrict__ bias,
                                                  __bf16* __restrict__ Qo,
                                                  __bf16* __restrict__ Ko,
                                                  __bf16* __restrict__ VTo) {
  const int wid = threadIdx.x >> 5;
  const int wr = wid & 3, wc = wid >> 2;
  const int m0 = blockIdx.x * 128 + wr * 32;
  const int j0 = blockIdx.y * 128 + wc * 64;
  v8f acc[2][4] = {};

  for (int k0 = 0; k0 < CDIM; k0 += 32) {
    if (k0 + 32 < CDIM) {
      __builtin_prefetch(X + (size_t)m0 * CDIM + k0 + 32, 0, 3);
      __builtin_prefetch(W + (size_t)j0 * CDIM + k0 + 32, 0, 3);
    }
    v16bf a0 = load_a16(X + (size_t)m0 * CDIM + k0, CDIM);
    v16bf a1 = load_a16(X + (size_t)(m0 + 16) * CDIM + k0, CDIM);
    v16bf b[4];
#pragma unroll
    for (int ni = 0; ni < 4; ++ni)
      b[ni] = load_b16(W + (size_t)(j0 + ni * 16) * CDIM + k0, CDIM);
#pragma unroll
    for (int ni = 0; ni < 4; ++ni) {
      acc[0][ni] = wmma_bf16(a0, b[ni], acc[0][ni]);
      acc[1][ni] = wmma_bf16(a1, b[ni], acc[1][ni]);
    }
  }

  const int lane = threadIdx.x & 31;
  const int cN = lane & 15, hi = lane >> 4;
#pragma unroll
  for (int mi = 0; mi < 2; ++mi)
#pragma unroll
    for (int ni = 0; ni < 4; ++ni)
#pragma unroll
      for (int e = 0; e < 8; ++e) {
        int m = m0 + mi * 16 + e + 8 * hi;
        int j = j0 + ni * 16 + cN;
        float val = acc[mi][ni][e] + bias[j];
        int which = j / CDIM;
        int c = j - which * CDIM;
        int h = c >> 6, d = c & 63;
        int bb = m >> 11, n = m & (SEQ - 1);
        size_t bh = (size_t)bb * HEADS + h;
        if (which == 0)
          Qo[(bh * SEQ + n) * DHEAD + d] = (__bf16)(val * 0.125f);
        else if (which == 1)
          Ko[(bh * SEQ + n) * DHEAD + d] = (__bf16)val;
        else
          VTo[(bh * DHEAD + d) * SEQ + n] = (__bf16)val;
      }
}

// ---------------------------------------------------------------------------
// Flash attention, transposed-score formulation, TDM-staged K/V tiles.
// Grid (B*H, SEQ/64); block 128 thr = 4 waves; wave owns 16 query columns.
// S^T = K_tile(16keys x 64d) x Q^T(64d x 16q)  -> per-lane softmax stats.
// O^T = sum_tiles  vT(16d x 32keys) x P^T(32keys x 16q).
// 32-key K/V tiles double-buffered in LDS via tensor_load_to_lds (wave 0
// issues; TENSORcnt + workgroup barrier synchronize).
// ---------------------------------------------------------------------------
__global__ __launch_bounds__(128) void attn_k(const __bf16* __restrict__ Q,
                                              const __bf16* __restrict__ K,
                                              const __bf16* __restrict__ VT,
                                              const float* __restrict__ pos,
                                              const float* __restrict__ slice_bias,
                                              __bf16* __restrict__ AO) {
  __shared__ float z_sh[SEQ];
  __shared__ float bias_sh[33];
  __shared__ __align__(16) __bf16 Ksh[2][32 * KP];  // keys x d(64), pitch 72
  __shared__ __align__(16) __bf16 Vsh[2][64 * VP];  // d x keys(32), pitch 40

  const int bh = blockIdx.x;
  const int b = bh / HEADS;
  const int h = bh - b * HEADS;

  for (int i = threadIdx.x; i < SEQ; i += 128)
    z_sh[i] = pos[((size_t)b * SEQ + i) * 3];
  if (threadIdx.x < 33)
    bias_sh[threadIdx.x] = slice_bias[h * 33 + threadIdx.x] * 0.5f;

  const int wid = threadIdx.x >> 5;
  const int q0 = blockIdx.y * 64 + wid * 16;
  const int lane = threadIdx.x & 31;
  const int cQ = lane & 15, hi = lane >> 4;

  const __bf16* Kbase = K + (size_t)bh * SEQ * DHEAD;
  const __bf16* Vbase = VT + (size_t)bh * DHEAD * SEQ;

  const unsigned kso = lds_off32(&Ksh[0][0]);
  const unsigned vso = lds_off32(&Vsh[0][0]);
  const unsigned kbufsz = 32 * KP * 2, vbufsz = 64 * VP * 2;

#if !USE_TDM
  // async fallback partitioning: 4 x b128 per wave per tile
  const int krow = (wid << 3) | (lane >> 3);
  const int kcol = (lane & 7) << 4;
  const int vrow = (wid << 4) | (lane >> 2);
  const int vcol = (lane & 3) << 4;
#endif

  auto issue_tile = [&](int buf, int t0) {
#if USE_TDM
    if (wid == 0) {
      tdm_load_ktile(kso + buf * kbufsz,
                     (unsigned long long)(uintptr_t)(Kbase + (size_t)t0 * DHEAD));
      tdm_load_vtile(vso + buf * vbufsz,
                     (unsigned long long)(uintptr_t)(Vbase + t0));
    }
#else
#pragma unroll
    for (int j = 0; j < 2; ++j) {
      int r = krow + j * 4;
      async_b128(kso + buf * kbufsz + (unsigned)(r * KP * 2 + kcol),
                 (const char*)(Kbase + (size_t)(t0 + r) * DHEAD) + kcol);
    }
#pragma unroll
    for (int j = 0; j < 2; ++j) {
      int r = vrow + j * 8;
      async_b128(vso + buf * vbufsz + (unsigned)(r * VP * 2 + vcol),
                 (const char*)(Vbase + (size_t)r * SEQ + t0) + vcol);
    }
#endif
  };

  const __bf16* Qb = Q + ((size_t)bh * SEQ + q0) * DHEAD;
  const v16bf qb0 = load_b16(Qb, DHEAD);       // d 0..31
  const v16bf qb1 = load_b16(Qb + 32, DHEAD);  // d 32..63

  issue_tile(0, 0);
  __syncthreads();  // covers z_sh/bias_sh staging too
  const float zq = z_sh[q0 + cQ];

  float m_st = -1e30f, l_st = 0.0f;
  v8f acc[4] = {};

  for (int it = 0; it < SEQ / 32; ++it) {
    const int t0 = it * 32;
    const int buf = it & 1;
    if (it + 1 < SEQ / 32) {
      issue_tile(buf ^ 1, t0 + 32);
      wait_stage_pipe();  // in-order: tile `it` has landed
    } else {
      wait_stage_all();
    }
    __syncthreads();  // all staged data visible to all waves

    const __bf16* kt = &Ksh[buf][0];
    const __bf16* vt = &Vsh[buf][0];

    v16bf ka00 = load_a16(kt, KP);                 // keys 0..15, d 0..31
    v16bf ka01 = load_a16(kt + 32, KP);            // keys 0..15, d 32..63
    v16bf ka10 = load_a16(kt + 16 * KP, KP);       // keys 16..31, d 0..31
    v16bf ka11 = load_a16(kt + 16 * KP + 32, KP);  // keys 16..31, d 32..63

    v8f s0 = {}, s1 = {};
    s0 = wmma_bf16(ka00, qb0, s0);
    s0 = wmma_bf16(ka01, qb1, s0);
    s1 = wmma_bf16(ka10, qb0, s1);
    s1 = wmma_bf16(ka11, qb1, s1);

    // add slice bias, find tile max (per query column = per lane)
    float sv0[8], sv1[8];
    float tmax = -1e30f;
#pragma unroll
    for (int e = 0; e < 8; ++e) {
      int k0i = t0 + e + 8 * hi;
      int k1i = k0i + 16;
      float zd0 = fabsf(zq - z_sh[k0i]);
      float zd1 = fabsf(zq - z_sh[k1i]);
      int i0 = (int)(zd0 * 32.0f); i0 = i0 > 32 ? 32 : i0;
      int i1 = (int)(zd1 * 32.0f); i1 = i1 > 32 ? 32 : i1;
      sv0[e] = s0[e] + bias_sh[i0];
      sv1[e] = s1[e] + bias_sh[i1];
      tmax = fmaxf(tmax, fmaxf(sv0[e], sv1[e]));
    }
    tmax = fmaxf(tmax, __shfl_xor(tmax, 16, 32));

    float m_new = fmaxf(m_st, tmax);
    float cf = __expf(m_st - m_new);
    float tsum = 0.0f;
    v8bf p0, p1;
#pragma unroll
    for (int e = 0; e < 8; ++e) {
      float e0 = __expf(sv0[e] - m_new);
      float e1 = __expf(sv1[e] - m_new);
      tsum += e0 + e1;
      p0[e] = (__bf16)e0;
      p1[e] = (__bf16)e1;
    }
    tsum += __shfl_xor(tsum, 16, 32);
    l_st = l_st * cf + tsum;
    m_st = m_new;

    // build P^T B-fragment (col=query, 16 contiguous keys per lane-half):
    // hi==0 wants keys t0+0..15  = [own p0 | partner p0]
    // hi==1 wants keys t0+16..31 = [partner p1 | own p1]
    union U { v8bf v; uint4 u; };
    U a0; a0.v = p0;
    U a1; a1.v = p1;
    U b0, b1;
    b0.u.x = __shfl_xor(a0.u.x, 16, 32);
    b0.u.y = __shfl_xor(a0.u.y, 16, 32);
    b0.u.z = __shfl_xor(a0.u.z, 16, 32);
    b0.u.w = __shfl_xor(a0.u.w, 16, 32);
    b1.u.x = __shfl_xor(a1.u.x, 16, 32);
    b1.u.y = __shfl_xor(a1.u.y, 16, 32);
    b1.u.z = __shfl_xor(a1.u.z, 16, 32);
    b1.u.w = __shfl_xor(a1.u.w, 16, 32);
    v16bf pB = hi ? concat8(b1.v, a1.v) : concat8(a0.v, b0.v);

#pragma unroll
    for (int df = 0; df < 4; ++df) {
      v16bf va = load_a16(vt + df * 16 * VP, VP);
      v8f c = acc[df];
#pragma unroll
      for (int e = 0; e < 8; ++e) c[e] *= cf;
      acc[df] = wmma_bf16(va, pB, c);
    }
    __syncthreads();  // tile fully consumed before buf is overwritten
  }

  const float inv_l = 1.0f / l_st;
  const int n = q0 + cQ;
#pragma unroll
  for (int df = 0; df < 4; ++df)
#pragma unroll
    for (int e = 0; e < 8; ++e) {
      int d = df * 16 + e + 8 * hi;
      AO[((size_t)b * SEQ + n) * CDIM + h * DHEAD + d] =
          (__bf16)(acc[df][e] * inv_l);
    }
}

// ---------------------------------------------------------------------------
// Output projection: [4096 x 768] x [768 x 768]^T + bias -> f32 out
// ---------------------------------------------------------------------------
__global__ __launch_bounds__(256) void proj_gemm_k(const __bf16* __restrict__ X,
                                                   const __bf16* __restrict__ W,
                                                   const float* __restrict__ bias,
                                                   float* __restrict__ Out) {
  const int wid = threadIdx.x >> 5;
  const int wr = wid & 3, wc = wid >> 2;
  const int m0 = blockIdx.x * 128 + wr * 32;
  const int j0 = blockIdx.y * 128 + wc * 64;
  v8f acc[2][4] = {};

  for (int k0 = 0; k0 < CDIM; k0 += 32) {
    if (k0 + 32 < CDIM) {
      __builtin_prefetch(X + (size_t)m0 * CDIM + k0 + 32, 0, 3);
      __builtin_prefetch(W + (size_t)j0 * CDIM + k0 + 32, 0, 3);
    }
    v16bf a0 = load_a16(X + (size_t)m0 * CDIM + k0, CDIM);
    v16bf a1 = load_a16(X + (size_t)(m0 + 16) * CDIM + k0, CDIM);
    v16bf b[4];
#pragma unroll
    for (int ni = 0; ni < 4; ++ni)
      b[ni] = load_b16(W + (size_t)(j0 + ni * 16) * CDIM + k0, CDIM);
#pragma unroll
    for (int ni = 0; ni < 4; ++ni) {
      acc[0][ni] = wmma_bf16(a0, b[ni], acc[0][ni]);
      acc[1][ni] = wmma_bf16(a1, b[ni], acc[1][ni]);
    }
  }

  const int lane = threadIdx.x & 31;
  const int cN = lane & 15, hi = lane >> 4;
#pragma unroll
  for (int mi = 0; mi < 2; ++mi)
#pragma unroll
    for (int ni = 0; ni < 4; ++ni)
#pragma unroll
      for (int e = 0; e < 8; ++e) {
        int m = m0 + mi * 16 + e + 8 * hi;
        int j = j0 + ni * 16 + cN;
        Out[(size_t)m * CDIM + j] = acc[mi][ni][e] + bias[j];
      }
}

// ---------------------------------------------------------------------------
extern "C" void kernel_launch(void* const* d_in, const int* in_sizes, int n_in,
                              void* d_out, int out_size, void* d_ws, size_t ws_size,
                              hipStream_t stream) {
  const float* x        = (const float*)d_in[0];  // (2,2048,768)
  const float* pos      = (const float*)d_in[1];  // (2,2048,3)
  const float* qkv_w    = (const float*)d_in[2];  // (2304,768)
  const float* qkv_b    = (const float*)d_in[3];  // (2304,)
  const float* proj_w   = (const float*)d_in[4];  // (768,768)
  const float* proj_b   = (const float*)d_in[5];  // (768,)
  const float* sbias    = (const float*)d_in[6];  // (12,33)
  float* out = (float*)d_out;

  char* ws = (char*)d_ws;
  size_t off = 0;
  auto alloc = [&](size_t bytes) {
    void* p = ws + off;
    off += (bytes + 255) & ~(size_t)255;
    return p;
  };
  const size_t n_x  = (size_t)BATCH * SEQ * CDIM;
  const size_t n_wq = (size_t)QKV_N * CDIM;
  const size_t n_wp = (size_t)CDIM * CDIM;
  const size_t n_h  = (size_t)BATCH * HEADS * SEQ * DHEAD;

  __bf16* x_bf  = (__bf16*)alloc(n_x * 2);
  __bf16* wq_bf = (__bf16*)alloc(n_wq * 2);
  __bf16* wp_bf = (__bf16*)alloc(n_wp * 2);
  __bf16* q_bf  = (__bf16*)alloc(n_h * 2);
  __bf16* k_bf  = (__bf16*)alloc(n_h * 2);
  __bf16* vT_bf = (__bf16*)alloc(n_h * 2);
  __bf16* ao_bf = (__bf16*)alloc(n_x * 2);

  f32_to_bf16_k<<<1024, 256, 0, stream>>>(x, x_bf, (int)n_x);
  f32_to_bf16_k<<<1024, 256, 0, stream>>>(qkv_w, wq_bf, (int)n_wq);
  f32_to_bf16_k<<<512, 256, 0, stream>>>(proj_w, wp_bf, (int)n_wp);

  qkv_gemm_k<<<dim3(32, 18), 256, 0, stream>>>(x_bf, wq_bf, qkv_b, q_bf, k_bf, vT_bf);

  attn_k<<<dim3(BATCH * HEADS, SEQ / 64), 128, 0, stream>>>(q_bf, k_bf, vT_bf, pos,
                                                            sbias, ao_bf);

  proj_gemm_k<<<dim3(32, 6), 256, 0, stream>>>(ao_bf, wp_bf, proj_b, out);
}
